// BaseResidualGNN_65644280152479
// MI455X (gfx1250) — compile-verified, matrix operand order checked by reading, not run on hardware
//
#include <hip/hip_runtime.h>
#include <stdint.h>

typedef __attribute__((ext_vector_type(16))) __bf16 bf16x16;
typedef __attribute__((ext_vector_type(8)))  float  floatx8;

__device__ __forceinline__ unsigned short f2bf(float f) {
  unsigned u = __float_as_uint(f);
  u = (u + 0x7FFFu + ((u >> 16) & 1u)) >> 16;   // round-to-nearest-even bf16
  return (unsigned short)u;
}

// wave-cooperative k-th largest |v| (radix-select over 32 lanes x 8 values)
__device__ __forceinline__ unsigned kth_threshold(const float* v8, int k) {
  unsigned v[8];
#pragma unroll
  for (int j = 0; j < 8; ++j) v[j] = __float_as_uint(fabsf(v8[j]));
  unsigned t = 0;
  for (int bit = 30; bit >= 0; --bit) {
    unsigned cand = t | (1u << bit);
    int c = 0;
#pragma unroll
    for (int j = 0; j < 8; ++j) c += (v[j] >= cand) ? 1 : 0;
    for (int off = 16; off; off >>= 1) c += __shfl_xor(c, off, 32);
    if (c >= k) t = cand;                  // max T with count(|v|>=T) >= k
  }
  return t;
}

__device__ __forceinline__ void ld8(const float* rowBase, int lane, float* d) {
  float4 a = ((const float4*)rowBase)[lane];
  float4 b = ((const float4*)rowBase)[lane + 32];
  d[0]=a.x; d[1]=a.y; d[2]=a.z; d[3]=a.w; d[4]=b.x; d[5]=b.y; d[6]=b.z; d[7]=b.w;
}
__device__ __forceinline__ void st8(float* rowBase, int lane, const float* d) {
  float4 a; a.x=d[0]; a.y=d[1]; a.z=d[2]; a.w=d[3];
  float4 b; b.x=d[4]; b.y=d[5]; b.z=d[6]; b.w=d[7];
  ((float4*)rowBase)[lane] = a;
  ((float4*)rowBase)[lane + 32] = b;
}
__device__ __forceinline__ void st8_bf16(unsigned short* rowBase, int lane, const float* d) {
  union { unsigned short u[4]; uint2 q; } a, b;
#pragma unroll
  for (int c = 0; c < 4; ++c) { a.u[c] = f2bf(d[c]); b.u[c] = f2bf(d[4 + c]); }
  ((uint2*)rowBase)[lane] = a.q;
  ((uint2*)rowBase)[lane + 32] = b.q;
}

// ---------------------------------------------------------------- utilities
__global__ __launch_bounds__(256) void fill_f32(float* p, float v, int count) {
  int i = blockIdx.x * blockDim.x + threadIdx.x;
  if (i < count) p[i] = v;
}
__global__ __launch_bounds__(256) void fill_u32(unsigned* p, unsigned v, int count) {
  int i = blockIdx.x * blockDim.x + threadIdx.x;
  if (i < count) p[i] = v;
}

// W [mats][K][256] fp32 -> Wt [mats][256][K] bf16 (N-major, contiguous in K)
__global__ __launch_bounds__(256) void wt_convert(const float* __restrict__ src,
                                                  unsigned short* __restrict__ dst,
                                                  int K, int total) {
  const int N = 256;
  int idx = blockIdx.x * blockDim.x + threadIdx.x;
  if (idx >= total) return;
  int m = idx / (K * N);
  int rem = idx - m * K * N;
  int nn = rem / K;
  int kk = rem - nn * K;
  dst[idx] = f2bf(src[(size_t)m * K * N + (size_t)kk * N + nn]);
}

__global__ __launch_bounds__(256) void f32_to_bf16(const float* __restrict__ src,
                                                   unsigned short* __restrict__ dst, int total) {
  int i = blockIdx.x * blockDim.x + threadIdx.x;
  if (i < total) dst[i] = f2bf(src[i]);
}

// ------------------------------------------------------------ CSR build
__global__ __launch_bounds__(256) void deg_accum_int(const int* __restrict__ cols,
                                                     int* __restrict__ cnt, int E) {
  int e = blockIdx.x * blockDim.x + threadIdx.x;
  if (e < E) atomicAdd(&cnt[cols[e]], 1);
}

__global__ __launch_bounds__(256) void make_dinv(const int* __restrict__ cnt,
                                                 float* __restrict__ dinv, int n) {
  int i = blockIdx.x * blockDim.x + threadIdx.x;
  if (i >= n) return;
  float d = (float)cnt[i];
  dinv[i] = d > 0.0f ? rsqrtf(d) : 0.0f;
}

// single-workgroup exclusive scan (1024 threads, LDS Hillis-Steele + carry)
__global__ __launch_bounds__(1024)
void scan_offsets(const int* __restrict__ cnt, int* __restrict__ offs, int n) {
  __shared__ int lds[1024];
  __shared__ int carry;
  if (threadIdx.x == 0) carry = 0;
  __syncthreads();
  for (int base = 0; base < n; base += 1024) {
    int i = base + (int)threadIdx.x;
    int v = (i < n) ? cnt[i] : 0;
    lds[threadIdx.x] = v;
    __syncthreads();
    for (int off = 1; off < 1024; off <<= 1) {
      int t = (threadIdx.x >= (unsigned)off) ? lds[threadIdx.x - off] : 0;
      __syncthreads();
      lds[threadIdx.x] += t;
      __syncthreads();
    }
    int incl = lds[threadIdx.x];
    int c = carry;
    if (i < n) offs[i] = c + incl - v;     // exclusive
    __syncthreads();
    if (threadIdx.x == 1023) carry = c + lds[1023];
    __syncthreads();
  }
  if (threadIdx.x == 0) offs[n] = carry;
}

__global__ __launch_bounds__(256)
void build_csr(const int* __restrict__ rows, const int* __restrict__ cols,
               const float* __restrict__ dinv, const int* __restrict__ offs,
               int* __restrict__ cursor, int* __restrict__ csr_src,
               float* __restrict__ csr_w, int E, int n) {
  int e = blockIdx.x * blockDim.x + threadIdx.x;
  if (e >= E + n) return;
  int r, c; float w;
  if (e < E) { r = rows[e]; c = cols[e]; w = dinv[r] * dinv[c]; }
  else { r = c = e - E; float d = dinv[r]; w = d * d; }   // self loop
  int pos = offs[c] + atomicAdd(&cursor[c], 1);
  csr_src[pos] = r;
  csr_w[pos] = w;
}

// ------------------------------------------------- bf16 WMMA GEMM: C = A @ B
// A: [M][K] bf16 row-major.  Bt: [256][K] bf16 (B transposed, N-major).
// One wave computes a 16x64 strip: 4 x v_wmma_f32_16x16x32_bf16 accumulators.
__global__ __launch_bounds__(256)
void gemm_wmma_bf16(const unsigned short* __restrict__ A,
                    const unsigned short* __restrict__ Bt,
                    float* __restrict__ C, int M, int K) {
  const int N = 256;
  int lane  = threadIdx.x & 31;
  int wave  = threadIdx.x >> 5;
  int mTile = blockIdx.x * 8 + wave;
  if (mTile * 16 >= M) return;
  int nBase = blockIdx.y * 64;
  int row   = mTile * 16 + (lane & 15);
  int khalf = (lane >> 4) << 3;          // lanes 0-15: K 0..7/16..23 ; 16-31: 8..15/24..31

  floatx8 acc[4] = {{0.f},{0.f},{0.f},{0.f}};

  for (int k0 = 0; k0 < K; k0 += 32) {
    const unsigned short* ap = A + (size_t)row * K + k0 + khalf;
    union { bf16x16 v; uint4 q[2]; } a;
    a.q[0] = *(const uint4*)(ap);        // K = base+0..7
    a.q[1] = *(const uint4*)(ap + 16);   // K = base+16..23
    if (k0 + 32 < K) __builtin_prefetch(ap + 32, 0, 0);   // global_prefetch_b8
#pragma unroll
    for (int j = 0; j < 4; ++j) {
      int colN = nBase + j * 16 + (lane & 15);
      const unsigned short* bp = Bt + (size_t)colN * K + k0 + khalf;
      union { bf16x16 v; uint4 q[2]; } b;
      b.q[0] = *(const uint4*)(bp);
      b.q[1] = *(const uint4*)(bp + 16);
      acc[j] = __builtin_amdgcn_wmma_f32_16x16x32_bf16(
          false, a.v, false, b.v, (short)0, acc[j], false, false);
    }
  }

  int mOff = mTile * 16 + ((lane >> 4) << 3);
#pragma unroll
  for (int j = 0; j < 4; ++j) {
    int colN = nBase + j * 16 + (lane & 15);
#pragma unroll
    for (int r = 0; r < 8; ++r)
      C[(size_t)(mOff + r) * N + colN] = acc[j][r];
  }
}

// ------------- wave-per-node CSR aggregation + bias + relu + row threshold
__global__ __launch_bounds__(256)
void aggregate_rows(const int* __restrict__ offs, const int* __restrict__ csr_src,
                    const float* __restrict__ csr_w, const float* __restrict__ hw,
                    const float* __restrict__ bias, float* __restrict__ out,
                    unsigned* __restrict__ thr, int n) {
  int g = blockIdx.x * blockDim.x + threadIdx.x;
  int i = g >> 5, lane = g & 31;
  if (i >= n) return;
  int beg = offs[i], end = offs[i + 1];
  float acc[8] = {0.f,0.f,0.f,0.f,0.f,0.f,0.f,0.f};
  for (int j = beg; j < end; ++j) {
    int s = csr_src[j];
    float w = csr_w[j];
    if (j + 1 < end) __builtin_prefetch(hw + (size_t)csr_src[j + 1] * 256, 0, 0);
    const float4* src = (const float4*)(hw + (size_t)s * 256);
    float4 v0 = src[lane], v1 = src[lane + 32];
    acc[0] += v0.x * w; acc[1] += v0.y * w; acc[2] += v0.z * w; acc[3] += v0.w * w;
    acc[4] += v1.x * w; acc[5] += v1.y * w; acc[6] += v1.z * w; acc[7] += v1.w * w;
  }
  float o[8];
#pragma unroll
  for (int c = 0; c < 4; ++c) {
    o[c]     = fmaxf(acc[c]     + bias[lane * 4 + c],       0.0f);
    o[4 + c] = fmaxf(acc[4 + c] + bias[128 + lane * 4 + c], 0.0f);
  }
  st8(out + (size_t)i * 256, lane, o);
  unsigned t = kth_threshold(o, 128);
  if (lane == 0) thr[i] = t;
}

// ---- wave-per-row fusion: gated masked residual + relu; writes fp32 + bf16
// states and their thresholds (in-place over props set); last layer pools.
__global__ __launch_bounds__(256)
void fuse_rows(float* p0, float* p1, float* p2,
               const unsigned* __restrict__ t0, const unsigned* __restrict__ t1,
               const unsigned* __restrict__ t2,
               const float* s0, const float* s1, const float* s2,
               unsigned* u0, unsigned* u1, unsigned* u2,
               unsigned short* q0, unsigned short* q1, unsigned short* q2,
               const float* __restrict__ gate_logit,
               const int* __restrict__ batch, float* __restrict__ pool,
               int n, int useStates, int last) {
  int g = blockIdx.x * blockDim.x + threadIdx.x;
  int i = g >> 5, lane = g & 31;
  if (i >= n) return;
  float gate = 1.0f / (1.0f + expf(-gate_logit[0]));
  size_t rb = (size_t)i * 256;

  float P0[8], P1[8], P2[8];
  ld8(p0 + rb, lane, P0); ld8(p1 + rb, lane, P1); ld8(p2 + rb, lane, P2);
  unsigned T0 = t0[i], T1 = t1[i], T2 = t2[i];

  float S0[8], S1[8], S2[8];
  unsigned U0 = 0, U1 = 0, U2 = 0;
  if (useStates) {
    ld8(s0 + rb, lane, S0); ld8(s1 + rb, lane, S1); ld8(s2 + rb, lane, S2);
    U0 = u0[i]; U1 = u1[i]; U2 = u2[i];
  }

  float O0[8], O1[8], O2[8];
#pragma unroll
  for (int j = 0; j < 8; ++j) {
    float m0 = (__float_as_uint(fabsf(P0[j])) >= T0) ? P0[j] : 0.0f;
    float m1 = (__float_as_uint(fabsf(P1[j])) >= T1) ? P1[j] : 0.0f;
    float m2 = (__float_as_uint(fabsf(P2[j])) >= T2) ? P2[j] : 0.0f;
    float r0 = m1, r1 = m0 + m2, r2 = m1;
    if (useStates) {
      float n0 = (__float_as_uint(fabsf(S0[j])) >= U0) ? S0[j] : 0.0f;
      float n1 = (__float_as_uint(fabsf(S1[j])) >= U1) ? S1[j] : 0.0f;
      float n2 = (__float_as_uint(fabsf(S2[j])) >= U2) ? S2[j] : 0.0f;
      r0 += n0 + n1;
      r1 += n1 + n0 + n2;
      r2 += n2 + n1;
    }
    O0[j] = fmaxf(P0[j] + gate * r0, 0.0f);
    O1[j] = fmaxf(P1[j] + gate * r1, 0.0f);
    O2[j] = fmaxf(P2[j] + gate * r2, 0.0f);
  }

  if (!last) {
    st8(p0 + rb, lane, O0); st8(p1 + rb, lane, O1); st8(p2 + rb, lane, O2);
    st8_bf16(q0 + rb, lane, O0); st8_bf16(q1 + rb, lane, O1); st8_bf16(q2 + rb, lane, O2);
    unsigned n0 = kth_threshold(O0, 128);
    unsigned n1 = kth_threshold(O1, 128);
    unsigned n2 = kth_threshold(O2, 128);
    if (lane == 0) { u0[i] = n0; u1[i] = n1; u2[i] = n2; }
  } else {
    const float inv3 = 1.0f / 3.0f;
    float* dst = pool + (size_t)batch[i] * 256;
#pragma unroll
    for (int c = 0; c < 4; ++c) {
      atomicAdd(dst + lane * 4 + c,       (O0[c] + O1[c] + O2[c]) * inv3);
      atomicAdd(dst + 128 + lane * 4 + c, (O0[4+c] + O1[4+c] + O2[4+c]) * inv3);
    }
  }
}

// ------------------------------------------------------------- pooling/head
__global__ __launch_bounds__(256) void count_nodes(const int* __restrict__ batch,
                                                   float* __restrict__ counts, int n) {
  int i = blockIdx.x * blockDim.x + threadIdx.x;
  if (i < n) atomicAdd(&counts[batch[i]], 1.0f);
}

__global__ __launch_bounds__(256)
void finalize_final(const float* __restrict__ pool, const float* __restrict__ counts,
                    float* __restrict__ outFinal) {
  int idx = blockIdx.x * blockDim.x + threadIdx.x;
  if (idx >= 256 * 256) return;
  outFinal[idx] = pool[idx] / fmaxf(counts[idx >> 8], 1.0f);
}

__global__ __launch_bounds__(256)
void logits_kernel(const float* __restrict__ finalBuf, const float* __restrict__ Wc,
                   const float* __restrict__ bc, float* __restrict__ outLogits) {
  int idx = blockIdx.x * blockDim.x + threadIdx.x;
  if (idx >= 256 * 10) return;
  int g = idx / 10, c = idx - g * 10;
  float s = bc[c];
  const float* f = finalBuf + g * 256;
  for (int j = 0; j < 256; ++j) s += f[j] * Wc[j * 10 + c];
  outLogits[idx] = s;
}

// --------------------------------------------------------------------- host
extern "C" void kernel_launch(void* const* d_in, const int* in_sizes, int n_in,
                              void* d_out, int out_size, void* d_ws, size_t ws_size,
                              hipStream_t stream) {
  const float* x    = (const float*)d_in[0];
  const int*   ei   = (const int*)d_in[1];
  const int*   batch= (const int*)d_in[2];
  const float* Win  = (const float*)d_in[3];
  const float* b_in = (const float*)d_in[4];
  const float* Whid = (const float*)d_in[5];
  const float* bhid = (const float*)d_in[6];
  const float* gate = (const float*)d_in[7];
  const float* Wc   = (const float*)d_in[8];
  const float* bc   = (const float*)d_in[9];

  const int n = in_sizes[2];
  const int E = in_sizes[1] / 2;
  const int IN = in_sizes[0] / n;       // 128
  const int* rows = ei;
  const int* cols = ei + E;
  const int H = 256, G = 256;

  // workspace carve-up
  char* base = (char*)d_ws;
  size_t off = 0;
  auto take = [&](size_t bytes) {
    size_t o = off;
    off += (bytes + 255) & ~(size_t)255;
    return base + o;
  };
  float*    dinv    = (float*)take((size_t)n * 4);
  int*      cnt     = (int*)take((size_t)n * 4);
  int*      cursor  = (int*)take((size_t)n * 4);
  int*      offs    = (int*)take((size_t)(n + 1) * 4);
  int*      csr_src = (int*)take((size_t)(E + n) * 4);
  float*    csr_w   = (float*)take((size_t)(E + n) * 4);
  float*    big[6];
  for (int i = 0; i < 6; ++i) big[i] = (float*)take((size_t)n * H * 4);
  float*    hw      = (float*)take((size_t)n * H * 4);
  unsigned short* xbf   = (unsigned short*)take((size_t)n * IN * 2);
  unsigned short* sbf   = (unsigned short*)take((size_t)3 * n * H * 2);
  unsigned short* wtin  = (unsigned short*)take((size_t)3 * IN * 256 * 2);
  unsigned short* wthid = (unsigned short*)take((size_t)9 * 256 * 256 * 2);
  unsigned* thrP  = (unsigned*)take((size_t)3 * n * 4);
  unsigned* thrS  = (unsigned*)take((size_t)3 * n * 4);
  float*    pool  = (float*)take((size_t)G * H * 4);
  float*    counts= (float*)take((size_t)G * 4);

  auto blocks = [](long long total) { return (unsigned)((total + 255) / 256); };

  // weights / input -> bf16
  wt_convert<<<blocks(3LL * IN * 256), 256, 0, stream>>>(Win,  wtin,  IN, 3 * IN * 256);
  wt_convert<<<blocks(9LL * 256 * 256), 256, 0, stream>>>(Whid, wthid, 256, 9 * 256 * 256);
  f32_to_bf16<<<blocks((long long)n * IN), 256, 0, stream>>>(x, xbf, n * IN);

  // CSR build: degree (with self loop) -> dinv -> exclusive scan -> scatter
  fill_u32<<<blocks(n), 256, 0, stream>>>((unsigned*)cnt, 1u, n);
  fill_u32<<<blocks(n), 256, 0, stream>>>((unsigned*)cursor, 0u, n);
  deg_accum_int<<<blocks(E), 256, 0, stream>>>(cols, cnt, E);
  make_dinv<<<blocks(n), 256, 0, stream>>>(cnt, dinv, n);
  scan_offsets<<<1, 1024, 0, stream>>>(cnt, offs, n);
  build_csr<<<blocks(E + n), 256, 0, stream>>>(rows, cols, dinv, offs, cursor,
                                               csr_src, csr_w, E, n);

  // pooling accumulators
  fill_f32<<<blocks(G * H), 256, 0, stream>>>(pool, 0.0f, G * H);
  fill_f32<<<1, 256, 0, stream>>>(counts, 0.0f, G);
  count_nodes<<<blocks(n), 256, 0, stream>>>(batch, counts, n);

  int pIdx = 0, sIdx = 1;               // ping-pong buffer-set roles
  const int mt = (n + 15) / 16;

  for (int L = 0; L < 4; ++L) {
    for (int b = 0; b < 3; ++b) {
      const unsigned short* Abf = (L == 0) ? xbf : sbf + (size_t)b * n * H;
      int K = (L == 0) ? IN : 256;
      const unsigned short* Bt = (L == 0)
          ? wtin + (size_t)b * IN * 256
          : wthid + (size_t)((L - 1) * 3 + b) * 256 * 256;
      gemm_wmma_bf16<<<dim3((mt + 7) / 8, 4), 256, 0, stream>>>(Abf, Bt, hw, n, K);
      const float* bias = (L == 0) ? b_in + b * H : bhid + ((L - 1) * 3 + b) * H;
      aggregate_rows<<<blocks((long long)n * 32), 256, 0, stream>>>(
          offs, csr_src, csr_w, hw, bias, big[pIdx * 3 + b], thrP + (size_t)b * n, n);
    }
    fuse_rows<<<blocks((long long)n * 32), 256, 0, stream>>>(
        big[pIdx * 3 + 0], big[pIdx * 3 + 1], big[pIdx * 3 + 2],
        thrP, thrP + n, thrP + 2 * (size_t)n,
        big[sIdx * 3 + 0], big[sIdx * 3 + 1], big[sIdx * 3 + 2],
        thrS, thrS + n, thrS + 2 * (size_t)n,
        sbf, sbf + (size_t)n * H, sbf + 2 * (size_t)n * H,
        gate, batch, pool, n, L > 0 ? 1 : 0, L == 3 ? 1 : 0);
    int t = pIdx; pIdx = sIdx; sIdx = t;
  }

  float* out = (float*)d_out;           // [logits 256*10 | final 256*256]
  finalize_final<<<blocks(G * H), 256, 0, stream>>>(pool, counts, out + G * 10);
  logits_kernel<<<blocks(G * 10), 256, 0, stream>>>(out + G * 10, Wc, bc, out);
}
